// SATLayer_75247827026131
// MI455X (gfx1250) — matrix operand order repaired and checked by reading, not compiled, not run on hardware
//
#include <hip/hip_runtime.h>

constexpr int cB = 8;
constexpr int cN = 128;
constexpr int cD = 256;

typedef __attribute__((ext_vector_type(16))) __bf16 v16bf;
typedef __attribute__((ext_vector_type(8)))  float  v8f;
typedef __attribute__((ext_vector_type(8)))  __bf16 bf16x8;

union Frag { v16bf v; bf16x8 h[2]; };

// A fragment (16x32 bf16, MxK): lane L holds row M = L&15; half h = L>>4 selects
// K chunks [kb + h*8, +8) and [kb + 16 + h*8, +8)  (ISA 7.12.2 16-bit A layout)
static __device__ inline v16bf load_a_frag(const __bf16* p /* row base + kb + h*8 */) {
  Frag f;
  f.h[0] = *(const bf16x8*)(p);
  f.h[1] = *(const bf16x8*)(p + 16);
  return f.v;
}
// B fragment (32x16 bf16, KxN): lane L holds col N = L&15; half h selects
// K chunks [kb + h*8, +8) and [kb + 16 + h*8, +8)
static __device__ inline v16bf load_b_frag(const __bf16* p /* row base + kb + h*16 */) {
  Frag f;
  const __bf16* q = p - ((threadIdx.x >> 4) & 1) * 8;
  f.h[0] = *(const bf16x8*)(q);
  f.h[1] = *(const bf16x8*)(q + 16);
  return f.v;
}

static __device__ inline v8f wmma_bf16(v16bf a, v16bf b, v8f c) {
  return __builtin_amdgcn_wmma_f32_16x16x32_bf16(false, a, false, b, (short)0, c, false, false);
}

// LDS byte offset of a __shared__ object (generic -> addrspace(3) -> u32)
static __device__ inline unsigned lds_u32(void* p) {
  return (unsigned)(unsigned long long)(__attribute__((address_space(3))) char*)(char*)p;
}

// ---------------- prep kernels ----------------
__global__ void cast_f32_to_bf16(const float* __restrict__ in, __bf16* __restrict__ out, int n) {
  int i = blockIdx.x * blockDim.x + threadIdx.x;
  if (i < n) out[i] = (__bf16)in[i];
}
__global__ void cast_i32_to_bf16(const int* __restrict__ in, __bf16* __restrict__ out, int n) {
  int i = blockIdx.x * blockDim.x + threadIdx.x;
  if (i < n) out[i] = (__bf16)(float)in[i];
}
// WT[e*D + d] = W[d*D + e]
__global__ void transpose_w_bf16(const float* __restrict__ W, __bf16* __restrict__ WT) {
  int i = blockIdx.x * blockDim.x + threadIdx.x; // i = e*cD + d
  int e = i >> 8, d = i & (cD - 1);
  WT[i] = (__bf16)W[d * cD + e];
}

// ---------------- projection: Out[n,e] = X[n,:] @ W[:,e] + bias[e]  (bf16 out) ----------------
__global__ void __launch_bounds__(32) proj_kernel(const __bf16* __restrict__ Xbf,
                                                  const __bf16* __restrict__ WT,
                                                  const float* __restrict__ bias,
                                                  __bf16* __restrict__ Out) {
  int idx = blockIdx.x;
  int et = idx & 15; idx >>= 4;      // cD/16 = 16
  int nt = idx & 7;  idx >>= 3;      // cN/16 = 8
  int b  = idx;                      // cB = 8
  int lane = threadIdx.x & 31, ln = lane & 15, h = lane >> 4;

  const __bf16* arow = WT + (et * 16 + ln) * cD + h * 8;
  const __bf16* brow = Xbf + (size_t)b * cN * cD + (nt * 16 + ln) * cD + h * 16;
  v8f acc = {};
#pragma unroll
  for (int ks = 0; ks < 8; ++ks)
    acc = wmma_bf16(load_a_frag(arow + ks * 32), load_b_frag(brow + ks * 32), acc);

  int ebase = et * 16 + h * 8;
  int n = nt * 16 + ln;
  bf16x8 o;
#pragma unroll
  for (int r = 0; r < 8; ++r) o[r] = (__bf16)(acc[r] + bias[ebase + r]);
  *(bf16x8*)(Out + ((size_t)b * cN + n) * cD + ebase) = o;
}

// ---------------- scores: S[j,k] = exp((Q[j,:]·K[k,:]) / 16)  (f32, row-major j,k) ----------------
__global__ void __launch_bounds__(32) scores_kernel(const __bf16* __restrict__ Qbf,
                                                    const __bf16* __restrict__ Kbf,
                                                    float* __restrict__ S) {
  int idx = blockIdx.x;
  int jt = idx & 7; idx >>= 3;
  int kt = idx & 7; idx >>= 3;
  int b  = idx;
  int lane = threadIdx.x & 31, ln = lane & 15, h = lane >> 4;

  const __bf16* arow = Kbf + (size_t)b * cN * cD + (kt * 16 + ln) * cD + h * 8;
  const __bf16* brow = Qbf + (size_t)b * cN * cD + (jt * 16 + ln) * cD + h * 16;
  v8f acc = {};
#pragma unroll
  for (int ks = 0; ks < 8; ++ks)
    acc = wmma_bf16(load_a_frag(arow + ks * 32), load_b_frag(brow + ks * 32), acc);

  int j = jt * 16 + ln;
  int kbase = kt * 16 + h * 8;
  float4 s0, s1;
  s0.x = __expf(acc[0] * 0.0625f); s0.y = __expf(acc[1] * 0.0625f);
  s0.z = __expf(acc[2] * 0.0625f); s0.w = __expf(acc[3] * 0.0625f);
  s1.x = __expf(acc[4] * 0.0625f); s1.y = __expf(acc[5] * 0.0625f);
  s1.z = __expf(acc[6] * 0.0625f); s1.w = __expf(acc[7] * 0.0625f);
  float* out = S + ((size_t)b * cN + j) * cN + kbase;
  *(float4*)(out) = s0;
  *(float4*)(out + 4) = s1;
}

// ---------------- voT[b, d, k] = (V @ Wo)^T in bf16 (contiguous along k) ----------------
__global__ void __launch_bounds__(32) vo_kernel(const __bf16* __restrict__ Vbf,
                                                const __bf16* __restrict__ WoT,
                                                __bf16* __restrict__ voT) {
  int idx = blockIdx.x;
  int dt = idx & 15; idx >>= 4;
  int kt = idx & 7;  idx >>= 3;
  int b  = idx;
  int lane = threadIdx.x & 31, ln = lane & 15, h = lane >> 4;

  const __bf16* arow = Vbf + (size_t)b * cN * cD + (kt * 16 + ln) * cD + h * 8;
  const __bf16* brow = WoT + (dt * 16 + ln) * cD + h * 16;
  v8f acc = {};
#pragma unroll
  for (int es = 0; es < 8; ++es)
    acc = wmma_bf16(load_a_frag(arow + es * 32), load_b_frag(brow + es * 32), acc);

  int d = dt * 16 + ln;
  int kbase = kt * 16 + h * 8;
  bf16x8 o;
#pragma unroll
  for (int r = 0; r < 8; ++r) o[r] = (__bf16)acc[r];
  *(bf16x8*)(voT + ((size_t)b * cD + d) * cN + kbase) = o;
}

// ---------------- main: per (b,j) WG: Num = adj_b @ diag(w_j) voT_b^T; masked max over i ----------------
__global__ void __launch_bounds__(256) main_kernel(const int* __restrict__ adj,
                                                   const float* __restrict__ S,
                                                   const __bf16* __restrict__ Abf,
                                                   const __bf16* __restrict__ voT,
                                                   const float* __restrict__ bo,
                                                   float* __restrict__ Out) {
  __shared__ float wsr[cN];
  __shared__ __align__(16) __bf16 wbf[cN];
  __shared__ float2 db[cN];
  __shared__ __align__(16) __bf16 Atile[cN * cN];   // 32 KB adjacency tile

  int b = blockIdx.x >> 7;
  int j = blockIdx.x & (cN - 1);
  int t = threadIdx.x;

  const __bf16* vob = voT + (size_t)b * cD * cN;

  // --- stage adjacency tile: 2048 x 16B async global->LDS transfers (ASYNCcnt) ---
  {
    unsigned long long gbase = (unsigned long long)(Abf + (size_t)b * cN * cN);
    unsigned lbase = lds_u32(&Atile[0]);
#pragma unroll
    for (int q = 0; q < 8; ++q) {
      unsigned voff = (unsigned)((q * 256 + t) * 16);
      unsigned laddr = lbase + voff;
      asm volatile("global_load_async_to_lds_b128 %0, %1, %2"
                   :: "v"(laddr), "v"(voff), "s"(gbase) : "memory");
    }
  }

  // warm L2/WGP$ for the voT rows this WG will stream (global_prefetch_b8)
  __builtin_prefetch(vob + (size_t)t * cN, 0, 0);

  if (t < cN) wsr[t] = S[((size_t)b * cN + j) * cN + t];
  __syncthreads();

  if (t < cN) {
    int i = t;
    const int* arow = adj + ((size_t)b * cN + i) * cN;
    const int4* a4 = (const int4*)arow;
    float den = 0.f;
#pragma unroll 4
    for (int k4 = 0; k4 < cN / 4; ++k4) {
      int4 a = a4[k4];
      den += (float)a.x * wsr[4 * k4 + 0] + (float)a.y * wsr[4 * k4 + 1]
           + (float)a.z * wsr[4 * k4 + 2] + (float)a.w * wsr[4 * k4 + 3];
    }
    bool valid = arow[j] != 0;
    float2 p;
    p.x = valid ? 1.f / fmaxf(den, 1e-9f) : 0.f;
    p.y = valid ? 0.f : -1e30f;
    db[i] = p;
    wbf[i] = (__bf16)wsr[i];
  }

  // all async LDS writes from this wave complete, then make visible WG-wide
  asm volatile("s_wait_asynccnt 0" ::: "memory");
  __syncthreads();

  int wave = t >> 5, lane = t & 31, ln = lane & 15, h = lane >> 4;

  // Hoist B fragments: 2 d-tiles x 4 k-steps, scaled by w[k]
  v16bf Bf[2][4];
#pragma unroll
  for (int ds = 0; ds < 2; ++ds) {
    int d = wave * 32 + ds * 16 + ln;
#pragma unroll
    for (int ks = 0; ks < 4; ++ks) {
      int kb = ks * 32 + h * 8;
      Frag f;
      f.h[0] = *(const bf16x8*)(vob + (size_t)d * cN + kb);
      f.h[1] = *(const bf16x8*)(vob + (size_t)d * cN + kb + 16);
#pragma unroll
      for (int e = 0; e < 8; ++e) {
        f.h[0][e] = (__bf16)((float)f.h[0][e] * (float)wbf[kb + e]);
        f.h[1][e] = (__bf16)((float)f.h[1][e] * (float)wbf[kb + 16 + e]);
      }
      Bf[ds][ks] = f.v;
    }
  }

  v8f m0, m1;
#pragma unroll
  for (int r = 0; r < 8; ++r) { m0[r] = -3e38f; m1[r] = -3e38f; }

  for (int it = 0; it < 8; ++it) {
    const __bf16* arow = &Atile[(it * 16 + ln) * cN + h * 8];   // ds_load_b128 x2 per frag
    v8f a0 = {}, a1 = {};
#pragma unroll
    for (int ks = 0; ks < 4; ++ks) {
      v16bf a = load_a_frag(arow + ks * 32);
      a0 = wmma_bf16(a, Bf[0][ks], a0);
      a1 = wmma_bf16(a, Bf[1][ks], a1);
    }
#pragma unroll
    for (int r = 0; r < 8; ++r) {
      float2 p = db[it * 16 + h * 8 + r];
      m0[r] = fmaxf(m0[r], a0[r] * p.x + p.y);
      m1[r] = fmaxf(m1[r], a1[r] * p.x + p.y);
    }
  }

  float M0 = m0[0], M1 = m1[0];
#pragma unroll
  for (int r = 1; r < 8; ++r) { M0 = fmaxf(M0, m0[r]); M1 = fmaxf(M1, m1[r]); }
  M0 = fmaxf(M0, __shfl_xor(M0, 16, 32));   // combine lane halves (wave32)
  M1 = fmaxf(M1, __shfl_xor(M1, 16, 32));

  if (h == 0) {
    int d0 = wave * 32 + ln, d1 = d0 + 16;
    float* orow = Out + ((size_t)b * cN + j) * cD;
    orow[d0] = M0 + bo[d0];
    orow[d1] = M1 + bo[d1];
  }
}

// ---------------- host launcher ----------------
extern "C" void kernel_launch(void* const* d_in, const int* in_sizes, int n_in,
                              void* d_out, int out_size, void* d_ws, size_t ws_size,
                              hipStream_t stream) {
  (void)in_sizes; (void)n_in; (void)out_size; (void)ws_size;
  const int*   adj = (const int*)d_in[0];
  const float* x   = (const float*)d_in[1];
  // d_in[2] = adj_mask (recomputed from adj, unused)
  const float* Wq = (const float*)d_in[3]; const float* bq = (const float*)d_in[4];
  const float* Wk = (const float*)d_in[5]; const float* bk = (const float*)d_in[6];
  const float* Wv = (const float*)d_in[7]; const float* bv = (const float*)d_in[8];
  const float* Wo = (const float*)d_in[9]; const float* bo = (const float*)d_in[10];
  float* out = (float*)d_out;

  char* w = (char*)d_ws;
  size_t off = 0;
  auto alloc = [&](size_t bytes) -> void* {
    void* p = w + off;
    off += (bytes + 255) & ~size_t(255);
    return p;
  };
  const size_t BND = (size_t)cB * cN * cD;   // 262144
  const size_t BNN = (size_t)cB * cN * cN;   // 131072
  const size_t DDm = (size_t)cD * cD;        // 65536

  __bf16* Xbf = (__bf16*)alloc(BND * 2);
  __bf16* WqT = (__bf16*)alloc(DDm * 2);
  __bf16* WkT = (__bf16*)alloc(DDm * 2);
  __bf16* WvT = (__bf16*)alloc(DDm * 2);
  __bf16* WoT = (__bf16*)alloc(DDm * 2);
  __bf16* Abf = (__bf16*)alloc(BNN * 2);
  __bf16* Qbf = (__bf16*)alloc(BND * 2);
  __bf16* Kbf = (__bf16*)alloc(BND * 2);
  __bf16* Vbf = (__bf16*)alloc(BND * 2);
  float*  S   = (float*) alloc(BNN * 4);
  __bf16* voT = (__bf16*)alloc(BND * 2);

  cast_f32_to_bf16<<<(int)((BND + 255) / 256), 256, 0, stream>>>(x, Xbf, (int)BND);
  cast_i32_to_bf16<<<(int)((BNN + 255) / 256), 256, 0, stream>>>(adj, Abf, (int)BNN);
  transpose_w_bf16<<<(int)(DDm / 256), 256, 0, stream>>>(Wq, WqT);
  transpose_w_bf16<<<(int)(DDm / 256), 256, 0, stream>>>(Wk, WkT);
  transpose_w_bf16<<<(int)(DDm / 256), 256, 0, stream>>>(Wv, WvT);
  transpose_w_bf16<<<(int)(DDm / 256), 256, 0, stream>>>(Wo, WoT);

  proj_kernel<<<1024, 32, 0, stream>>>(Xbf, WqT, bq, Qbf);
  proj_kernel<<<1024, 32, 0, stream>>>(Xbf, WkT, bk, Kbf);
  proj_kernel<<<1024, 32, 0, stream>>>(Xbf, WvT, bv, Vbf);

  scores_kernel<<<512, 32, 0, stream>>>(Qbf, Kbf, S);
  vo_kernel<<<1024, 32, 0, stream>>>(Vbf, WoT, voT);

  main_kernel<<<cB * cN, 256, 0, stream>>>(adj, S, Abf, voT, bo, out);
}
